// LSTM_GraphTransformer_52604759441726
// MI455X (gfx1250) — matrix operand order repaired and checked by reading, not compiled, not run on hardware
//
#include <hip/hip_runtime.h>
#include <hip/hip_bf16.h>
#include <math.h>

// ---------------------------------------------------------------------------
// Problem constants
// ---------------------------------------------------------------------------
#define BB   16
#define TT   128
#define NN   200
#define EE   6400
#define HH   128
#define DD   264
#define DP   288          // padded D (multiple of 32)
#define NTOT 3200         // B*N nodes
#define EBT  102400       // B*E edges
#define CPH  66           // channels per head (264/4)

typedef __attribute__((ext_vector_type(16))) _Float16 v16h;
typedef __attribute__((ext_vector_type(8)))  float    v8f;

#define WMMA_F16(a,b,c) __builtin_amdgcn_wmma_f32_16x16x32_f16(false,(a),false,(b),(short)0,(c),false,false)

// Async global->LDS copy of 16 bytes (CDNA5 GLOBAL_LOAD_ASYNC_TO_LDS_B128,
// tracked by ASYNCcnt).  ldsoff = LDS byte offset (low 32 bits of generic
// shared pointer), gaddr = 64-bit global address.
__device__ __forceinline__ void async_g2l_b128(unsigned ldsoff, unsigned long long gaddr) {
  asm volatile("global_load_async_to_lds_b128 %0, %1, off"
               :: "v"(ldsoff), "v"(gaddr) : "memory");
}
__device__ __forceinline__ void wait_asynccnt0() {
  asm volatile("s_wait_asynccnt 0" ::: "memory");
}

// ---------------------------------------------------------------------------
// WMMA fragment helpers (wave32 16x16x32 f16; A and B both use contiguous k = 16*(lane>>4) + i)
// ---------------------------------------------------------------------------
__device__ __forceinline__ v16h load_A_frag(const _Float16* __restrict__ p0, int lda) {
  int lane = threadIdx.x & 31;
  const _Float16* p = p0 + (size_t)(lane & 15) * lda + ((lane >> 4) << 4);
  v16h a;
#pragma unroll
  for (int i = 0; i < 16; ++i) a[i] = p[i];
  return a;
}

// B 32x16 built from row-major W[N][K]: B[k][n] = W[n][k].
__device__ __forceinline__ v16h load_B_frag(const _Float16* __restrict__ p0, int ldw) {
  int lane = threadIdx.x & 31;
  const _Float16* p = p0 + (size_t)(lane & 15) * ldw + ((lane >> 4) << 4);
  v16h b;
#pragma unroll
  for (int i = 0; i < 16; ++i) b[i] = p[i];
  return b;
}

// C/D 16x16 f32: VGPR r -> row r + 8*(lane>>4), col = lane&15
__device__ __forceinline__ void store_C_frag(float* __restrict__ C, int ldc, v8f acc) {
  int lane = threadIdx.x & 31;
  float* p = C + (size_t)((lane >> 4) << 3) * ldc + (lane & 15);
#pragma unroll
  for (int r = 0; r < 8; ++r) p[(size_t)r * ldc] = acc[r];
}

__device__ __forceinline__ float sigf(float x) { return 1.f / (1.f + expf(-x)); }

__device__ __forceinline__ void atomicMaxFloat(float* addr, float val) {
  unsigned int* ua = (unsigned int*)addr;
  unsigned int old = *ua;
  while (__uint_as_float(old) < val) {
    unsigned int assumed = old;
    old = atomicCAS(ua, assumed, __float_as_uint(val));
    if (old == assumed) break;
  }
}

// ---------------------------------------------------------------------------
// Generic utility kernels
// ---------------------------------------------------------------------------
__global__ void k_fill_f32(float* p, long n, float v) {
  long i = (long)blockIdx.x * blockDim.x + threadIdx.x;
  long st = (long)gridDim.x * blockDim.x;
  for (; i < n; i += st) p[i] = v;
}

__global__ void k_f32_to_f16(_Float16* dst, const float* src, long n) {
  long i = (long)blockIdx.x * blockDim.x + threadIdx.x;
  long st = (long)gridDim.x * blockDim.x;
  for (; i < n; i += st) dst[i] = (_Float16)src[i];
}

// pad 264x264 f32 weight -> 288x288 f16 (zeros elsewhere)
__global__ void k_pad_w(_Float16* dst, const float* src) {
  long i = (long)blockIdx.x * blockDim.x + threadIdx.x;
  long st = (long)gridDim.x * blockDim.x;
  for (; i < (long)DP * DP; i += st) {
    int n = (int)(i / DP), k = (int)(i % DP);
    dst[i] = (n < DD && k < DD) ? (_Float16)src[(size_t)n * DD + k] : (_Float16)0.f;
  }
}

// ---------------------------------------------------------------------------
// PE: adjacency, Laplacian, Jacobi eigensolver, sorted eigvec extraction
// ---------------------------------------------------------------------------
__global__ void k_build_graph(const int* ei, float* A, float* deg) {
  int e = blockIdx.x * blockDim.x + threadIdx.x;
  if (e >= EE) return;
  int s = ei[e], d = ei[EE + e];
  atomicAdd(&A[(size_t)s * NN + d], 1.f);
  atomicAdd(&deg[s], 1.f);
}

__global__ void k_build_L(float* A, const float* deg) {
  int idx = blockIdx.x * blockDim.x + threadIdx.x;
  if (idx >= NN * NN) return;
  int i = idx / NN, j = idx % NN;
  float di = deg[i] > 0.f ? rsqrtf(deg[i]) : 0.f;
  float dj = deg[j] > 0.f ? rsqrtf(deg[j]) : 0.f;
  A[idx] = (i == j ? 1.f : 0.f) - di * A[idx] * dj;
}

__global__ void k_jacobi(float* A, float* V, int n, int sweeps) {
  __shared__ float cs[2];
  __shared__ int doRot;
  int tid = threadIdx.x;
  for (int i = tid; i < n * n; i += blockDim.x) V[i] = (i / n == i % n) ? 1.f : 0.f;
  __syncthreads();
  for (int sw = 0; sw < sweeps; ++sw) {
    for (int p = 0; p < n - 1; ++p) {
      for (int q = p + 1; q < n; ++q) {
        if (tid == 0) {
          float apq = A[p * n + q];
          if (fabsf(apq) > 1e-10f) {
            float app = A[p * n + p], aqq = A[q * n + q];
            float tau = (aqq - app) / (2.f * apq);
            float t = ((tau >= 0.f) ? 1.f : -1.f) / (fabsf(tau) + sqrtf(1.f + tau * tau));
            float c = rsqrtf(1.f + t * t);
            cs[0] = c; cs[1] = t * c; doRot = 1;
          } else doRot = 0;
        }
        __syncthreads();
        if (doRot) {
          float c = cs[0], s = cs[1];
          for (int j = tid; j < n; j += blockDim.x) {
            float ap = A[p * n + j], aq = A[q * n + j];
            A[p * n + j] = c * ap - s * aq;
            A[q * n + j] = s * ap + c * aq;
          }
          __syncthreads();
          for (int j = tid; j < n; j += blockDim.x) {
            float ap = A[j * n + p], aq = A[j * n + q];
            A[j * n + p] = c * ap - s * aq;
            A[j * n + q] = s * ap + c * aq;
            float vp = V[j * n + p], vq = V[j * n + q];
            V[j * n + p] = c * vp - s * vq;
            V[j * n + q] = s * vp + c * vq;
          }
        }
        __syncthreads();
      }
    }
  }
}

__global__ void k_pe_extract(const float* A, const float* V, float* pe) {
  __shared__ float ev[256];
  __shared__ int rank[256];
  __shared__ int sel[8];
  int tid = threadIdx.x;
  if (tid < NN) ev[tid] = A[(size_t)tid * NN + tid];
  __syncthreads();
  if (tid < NN) {
    int r = 0;
    for (int j = 0; j < NN; ++j) {
      float e = ev[j];
      if (e < ev[tid] || (e == ev[tid] && j < tid)) r++;
    }
    rank[tid] = r;
  }
  __syncthreads();
  if (tid < NN && rank[tid] >= 1 && rank[tid] <= 8) sel[rank[tid] - 1] = tid;
  __syncthreads();
  for (int idx = tid; idx < NN * 8; idx += blockDim.x) {
    int row = idx / 8, col = idx % 8;
    pe[idx] = V[(size_t)row * NN + sel[col]];
  }
  __syncthreads();
  if (tid < 8) {
    float best = -1.f; int bi = 0;
    for (int row = 0; row < NN; ++row) {
      float a = fabsf(pe[row * 8 + tid]);
      if (a > best) { best = a; bi = row; }
    }
    float s = (pe[bi * 8 + tid] < 0.f) ? -1.f : 1.f;
    for (int row = 0; row < NN; ++row) pe[row * 8 + tid] *= s;
  }
}

// ---------------------------------------------------------------------------
// LSTM: one workgroup = 16 sequences, loops all T steps internally.
// gates(16x512) = Xt @ Wih^T + H @ Whh^T + b   via v_wmma_f32_16x16x32_f16
// Xt tile is staged into LDS with async global->LDS copies overlapping the
// hidden-state WMMAs (ASYNCcnt path).
// ---------------------------------------------------------------------------
__global__ void __launch_bounds__(256)
k_lstm(const float* __restrict__ x0, const _Float16* __restrict__ xin,
       const float* __restrict__ wih32, const _Float16* __restrict__ wih16,
       const _Float16* __restrict__ whh16,
       const float* __restrict__ bih, const float* __restrict__ bhh,
       _Float16* __restrict__ out, int in_size, int reverse, int dirofs) {
  __shared__ _Float16 hT[16][128];
  __shared__ _Float16 xT[16][256];
  __shared__ float gS[16][512];
  __shared__ float cS[16][128];
  int tid = threadIdx.x;
  int seq0 = blockIdx.x * 16;
  for (int i = tid; i < 16 * 128; i += 256) {
    hT[i >> 7][i & 127] = (_Float16)0.f;
    cS[i >> 7][i & 127] = 0.f;
  }
  __syncthreads();
  int lane = tid & 31, wv = tid >> 5;

  for (int step = 0; step < TT; ++step) {
    int t = reverse ? (TT - 1 - step) : step;
    // --- issue async staging of x_t tile (16 x 256 f16 = 8KB, 512 x 16B) ---
    if (in_size == 256) {
#pragma unroll
      for (int c = tid; c < 512; c += 256) {
        int r = c >> 5, seg = c & 31;
        unsigned ldsoff = (unsigned)(unsigned long long)&xT[r][seg * 8];
        unsigned long long ga =
            (unsigned long long)(xin + ((size_t)(seq0 + r) * TT + t) * 256 + seg * 8);
        async_g2l_b128(ldsoff, ga);
      }
    }
    // --- hidden-state WMMAs (overlap with async copy) ---
    v8f acc[4];
#pragma unroll
    for (int nt = 0; nt < 4; ++nt) {
      int n0 = (wv * 4 + nt) * 16;
      int col = n0 + (lane & 15);
      float bv = bih[col] + bhh[col];
#pragma unroll
      for (int r = 0; r < 8; ++r) acc[nt][r] = bv;
#pragma unroll
      for (int kk = 0; kk < 4; ++kk) {
        v16h af = load_A_frag(&hT[0][kk * 32], 128);
        v16h bf = load_B_frag(whh16 + (size_t)n0 * 128 + kk * 32, 128);
        acc[nt] = WMMA_F16(af, bf, acc[nt]);
      }
    }
    // --- input WMMAs from the staged LDS tile ---
    if (in_size == 256) {
      wait_asynccnt0();
      __syncthreads();
#pragma unroll
      for (int nt = 0; nt < 4; ++nt) {
        int n0 = (wv * 4 + nt) * 16;
#pragma unroll
        for (int kk = 0; kk < 8; ++kk) {
          v16h af = load_A_frag(&xT[0][kk * 32], 256);
          v16h bf = load_B_frag(wih16 + (size_t)n0 * 256 + kk * 32, 256);
          acc[nt] = WMMA_F16(af, bf, acc[nt]);
        }
      }
    }
    // --- store gates ---
    {
      int rb = (lane >> 4) << 3, c0 = lane & 15;
#pragma unroll
      for (int nt = 0; nt < 4; ++nt) {
        int n0 = (wv * 4 + nt) * 16;
#pragma unroll
        for (int r = 0; r < 8; ++r) gS[rb + r][n0 + c0] = acc[nt][r];
      }
    }
    __syncthreads();
    // --- pointwise LSTM cell update ---
    for (int idx = tid; idx < 16 * 128; idx += 256) {
      int r = idx >> 7, j = idx & 127;
      float gi = gS[r][j], gf = gS[r][128 + j], gg = gS[r][256 + j], go = gS[r][384 + j];
      if (in_size == 0) {
        int s = seq0 + r;
        int bb = s / NN, nn2 = s % NN;
        float xv = x0[((size_t)bb * TT + t) * NN + nn2];
        gi += xv * wih32[j];       gf += xv * wih32[128 + j];
        gg += xv * wih32[256 + j]; go += xv * wih32[384 + j];
      }
      float c = sigf(gf) * cS[r][j] + sigf(gi) * tanhf(gg);
      float h = sigf(go) * tanhf(c);
      cS[r][j] = c;
      _Float16 h16 = (_Float16)h;
      hT[r][j] = h16;
      out[((size_t)(seq0 + r) * TT + t) * 256 + dirofs + j] = h16;
    }
    __syncthreads();
  }
}

// ---------------------------------------------------------------------------
// node = [mean_t(act), pe, zeros] -> (3200 x 288) f32
// ---------------------------------------------------------------------------
__global__ void k_node(const _Float16* __restrict__ act, const float* __restrict__ pe,
                       float* __restrict__ node) {
  long i = (long)blockIdx.x * blockDim.x + threadIdx.x;
  long st = (long)gridDim.x * blockDim.x;
  for (; i < (long)NTOT * DP; i += st) {
    int s = (int)(i / DP), d = (int)(i % DP);
    float v = 0.f;
    if (d < 256) {
      const _Float16* p = act + (size_t)s * TT * 256 + d;
      float acc = 0.f;
      for (int t = 0; t < TT; ++t) acc += (float)p[(size_t)t * 256];
      v = acc * (1.f / TT);
    } else if (d < DD) {
      v = pe[(s % NN) * 8 + (d - 256)];
    }
    node[i] = v;
  }
}

// ---------------------------------------------------------------------------
// GEMM: C(Mx288) = A(Mx288 f16) @ W^T (288x288 f16 row-major N x K) + bias
// 4 waves per block, one 16x16 C tile per wave.  The block's 4 weight tiles
// (64 rows x 288 halves = 36KB) are staged into LDS with async copies.
// ---------------------------------------------------------------------------
__global__ void __launch_bounds__(128)
k_gemm288(const _Float16* __restrict__ A, const _Float16* __restrict__ W,
          const float* __restrict__ bias, int NV, float* __restrict__ C) {
  __shared__ _Float16 sW[4 * 16 * DP];
  int tid = threadIdx.y * 32 + threadIdx.x;
  int ntile0 = blockIdx.x * 4;
  // stage W rows [ntile0*16, ntile0*16+64) in 16B chunks (36 per row)
  for (int c = tid; c < 4 * 16 * (DP / 8); c += 128) {
    int row = c / (DP / 8);
    int seg = c % (DP / 8);
    int grow = ntile0 * 16 + row;
    if (grow < DP) {
      unsigned ldsoff = (unsigned)(unsigned long long)&sW[(size_t)row * DP + seg * 8];
      unsigned long long ga =
          (unsigned long long)(W + (size_t)grow * DP + seg * 8);
      async_g2l_b128(ldsoff, ga);
    }
  }
  wait_asynccnt0();
  __syncthreads();

  int ntile = ntile0 + threadIdx.y;
  if (ntile >= DP / 16) return;
  int m0 = blockIdx.y * 16;
  int n0 = ntile * 16;
  int lane = threadIdx.x & 31;
  int col = n0 + (lane & 15);
  float bv = (bias && col < NV) ? bias[col] : 0.f;
  v8f acc;
#pragma unroll
  for (int r = 0; r < 8; ++r) acc[r] = bv;
  const _Float16* Ab = A + (size_t)m0 * DP;
  const _Float16* Wb = &sW[(size_t)threadIdx.y * 16 * DP];
#pragma unroll
  for (int k0 = 0; k0 < DP; k0 += 32) {
    __builtin_prefetch(Ab + k0 + 64, 0, 1);
    v16h af = load_A_frag(Ab + k0, DP);
    v16h bf = load_B_frag(Wb + k0, DP);
    acc = WMMA_F16(af, bf, acc);
  }
  store_C_frag(C + (size_t)m0 * DP + n0, DP, acc);
}

// ---------------------------------------------------------------------------
// TransformerConv attention passes (per global edge g, head h)
// ---------------------------------------------------------------------------
__global__ void k_att1(const float* __restrict__ q, const float* __restrict__ kk,
                       const int* __restrict__ ei, const float* __restrict__ ew,
                       const float* __restrict__ we,
                       float* __restrict__ alpha, float* __restrict__ amax) {
  int idx = blockIdx.x * blockDim.x + threadIdx.x;
  if (idx >= EBT * 4) return;
  int g = idx >> 2, h = idx & 3;
  int b = g / EE, e0 = g % EE;
  int src = ei[e0] + b * NN, dst = ei[EE + e0] + b * NN;
  float w = ew[e0];
  const float* qp = q + (size_t)dst * DP + h * CPH;
  const float* kp = kk + (size_t)src * DP + h * CPH;
  const float* wp = we + h * CPH;
  float s = 0.f;
  for (int c = 0; c < CPH; ++c) s += qp[c] * (kp[c] + w * wp[c]);
  float a = s * 0.123091490979f;  // 1/sqrt(66)
  alpha[idx] = a;
  atomicMaxFloat(&amax[dst * 4 + h], a);
}

__global__ void k_att2(const int* __restrict__ ei, const float* __restrict__ amax,
                       float* __restrict__ alpha, float* __restrict__ denom) {
  int idx = blockIdx.x * blockDim.x + threadIdx.x;
  if (idx >= EBT * 4) return;
  int g = idx >> 2, h = idx & 3;
  int b = g / EE, e0 = g % EE;
  int dst = ei[EE + e0] + b * NN;
  float am = amax[dst * 4 + h];
  if (am <= -1e29f) am = 0.f;
  float ex = expf(alpha[idx] - am);
  alpha[idx] = ex;
  atomicAdd(&denom[dst * 4 + h], ex);
}

__global__ void k_att3(const int* __restrict__ ei, const float* __restrict__ ew,
                       const float* __restrict__ we, const float* __restrict__ v,
                       const float* __restrict__ alpha, const float* __restrict__ denom,
                       float* __restrict__ hnew) {
  int idx = blockIdx.x * blockDim.x + threadIdx.x;
  if (idx >= EBT * 4) return;
  int g = idx >> 2, h = idx & 3;
  int b = g / EE, e0 = g % EE;
  int src = ei[e0] + b * NN, dst = ei[EE + e0] + b * NN;
  float w = ew[e0];
  float a = alpha[idx] / (denom[dst * 4 + h] + 1e-16f);
  const float* vp = v + (size_t)src * DP + h * CPH;
  const float* wp = we + h * CPH;
  float* op = hnew + (size_t)dst * DP + h * CPH;
  for (int c = 0; c < CPH; ++c) atomicAdd(&op[c], (vp[c] + w * wp[c]) * a);
}

// ---------------------------------------------------------------------------
// GraphNorm + residual ReLU
// ---------------------------------------------------------------------------
__global__ void k_gn_mean(const float* __restrict__ h, float* __restrict__ mean) {
  int idx = blockIdx.x * blockDim.x + threadIdx.x;
  if (idx >= BB * DP) return;
  int b = idx / DP, d = idx % DP;
  float s = 0.f;
  for (int n = 0; n < NN; ++n) s += h[(size_t)(b * NN + n) * DP + d];
  mean[idx] = s * (1.f / NN);
}

__global__ void k_gn_center(const float* __restrict__ h, const float* __restrict__ mean,
                            const float* __restrict__ ms, float* __restrict__ cent) {
  long i = (long)blockIdx.x * blockDim.x + threadIdx.x;
  long st = (long)gridDim.x * blockDim.x;
  for (; i < (long)NTOT * DP; i += st) {
    int s = (int)(i / DP), d = (int)(i % DP);
    int b = s / NN;
    float m = (d < DD) ? ms[d] : 0.f;
    cent[i] = h[i] - m * mean[b * DP + d];
  }
}

__global__ void k_gn_var(const float* __restrict__ cent, float* __restrict__ var) {
  int idx = blockIdx.x * blockDim.x + threadIdx.x;
  if (idx >= BB * DP) return;
  int b = idx / DP, d = idx % DP;
  float s = 0.f;
  for (int n = 0; n < NN; ++n) {
    float v = cent[(size_t)(b * NN + n) * DP + d];
    s += v * v;
  }
  var[idx] = s * (1.f / NN);
}

__global__ void k_gn_final(const float* __restrict__ cent, const float* __restrict__ var,
                           const float* __restrict__ wgt, const float* __restrict__ bias,
                           const float* __restrict__ hin, float* __restrict__ hout) {
  long i = (long)blockIdx.x * blockDim.x + threadIdx.x;
  long st = (long)gridDim.x * blockDim.x;
  for (; i < (long)NTOT * DP; i += st) {
    int s = (int)(i / DP), d = (int)(i % DP);
    float v = 0.f;
    if (d < DD) {
      int b = s / NN;
      float nrm = wgt[d] * cent[i] * rsqrtf(var[b * DP + d] + 1e-5f) + bias[d];
      v = nrm + hin[i];
      v = v > 0.f ? v : 0.f;
    }
    hout[i] = v;
  }
}

// ---------------------------------------------------------------------------
// Pool over nodes + classifier
// ---------------------------------------------------------------------------
__global__ void k_pool_cls(const float* __restrict__ h, const float* __restrict__ cls_w,
                           const float* __restrict__ cls_b, float* __restrict__ out) {
  int b = blockIdx.x;
  __shared__ float red[256];
  float s = 0.f;
  for (int d = threadIdx.x; d < DD; d += 256) {
    float acc = 0.f;
    for (int n = 0; n < NN; ++n) acc += h[(size_t)(b * NN + n) * DP + d];
    s += (acc * (1.f / NN)) * cls_w[d];
  }
  red[threadIdx.x] = s;
  __syncthreads();
  for (int off = 128; off > 0; off >>= 1) {
    if (threadIdx.x < off) red[threadIdx.x] += red[threadIdx.x + off];
    __syncthreads();
  }
  if (threadIdx.x == 0) out[b] = red[0] + cls_b[0];
}

// ---------------------------------------------------------------------------
// Host
// ---------------------------------------------------------------------------
static inline char* bump(char*& p, size_t bytes) {
  char* r = p;
  p += (bytes + 255) & ~(size_t)255;
  return r;
}

extern "C" void kernel_launch(void* const* d_in, const int* in_sizes, int n_in,
                              void* d_out, int out_size, void* d_ws, size_t ws_size,
                              hipStream_t stream) {
  (void)in_sizes; (void)n_in; (void)out_size; (void)ws_size;
  // ---- input mapping (jax pytree order: dict keys sorted) ----
  const float* x   = (const float*)d_in[0];
  const int*   ei  = (const int*)d_in[1];
  const float* ew  = (const float*)d_in[2];
  const float* cls_b = (const float*)d_in[3];
  const float* cls_w = (const float*)d_in[4];
  const float *g_bk[3], *g_bq[3], *g_bskip[3], *g_bv[3], *g_we[3];
  const float *g_wk[3], *g_wq[3], *g_wskip[3], *g_wv[3];
  for (int l = 0; l < 3; ++l) {
    int b5 = 5 + 9 * l;
    g_bk[l]    = (const float*)d_in[b5 + 0];
    g_bq[l]    = (const float*)d_in[b5 + 1];
    g_bskip[l] = (const float*)d_in[b5 + 2];
    g_bv[l]    = (const float*)d_in[b5 + 3];
    g_we[l]    = (const float*)d_in[b5 + 4];
    g_wk[l]    = (const float*)d_in[b5 + 5];
    g_wq[l]    = (const float*)d_in[b5 + 6];
    g_wskip[l] = (const float*)d_in[b5 + 7];
    g_wv[l]    = (const float*)d_in[b5 + 8];
  }
  // lstm: per layer {bwd, fwd} each {b_hh, b_ih, w_hh, w_ih}; dir index 0=fwd,1=bwd
  const float *l_bhh[3][2], *l_bih[3][2], *l_whh[3][2], *l_wih[3][2];
  for (int l = 0; l < 3; ++l) {
    int base = 32 + 8 * l;
    l_bhh[l][1] = (const float*)d_in[base + 0];
    l_bih[l][1] = (const float*)d_in[base + 1];
    l_whh[l][1] = (const float*)d_in[base + 2];
    l_wih[l][1] = (const float*)d_in[base + 3];
    l_bhh[l][0] = (const float*)d_in[base + 4];
    l_bih[l][0] = (const float*)d_in[base + 5];
    l_whh[l][0] = (const float*)d_in[base + 6];
    l_wih[l][0] = (const float*)d_in[base + 7];
  }
  const float *n_bias[3], *n_ms[3], *n_w[3];
  for (int l = 0; l < 3; ++l) {
    int base = 56 + 3 * l;
    n_bias[l] = (const float*)d_in[base + 0];
    n_ms[l]   = (const float*)d_in[base + 1];
    n_w[l]    = (const float*)d_in[base + 2];
  }
  float* out = (float*)d_out;

  // ---- workspace carve ----
  char* p = (char*)d_ws;
  _Float16* actA  = (_Float16*)bump(p, (size_t)NTOT * TT * 256 * 2);
  _Float16* actB  = (_Float16*)bump(p, (size_t)NTOT * TT * 256 * 2);
  _Float16* whh16 = (_Float16*)bump(p, (size_t)6 * 512 * 128 * 2);
  _Float16* wih16 = (_Float16*)bump(p, (size_t)4 * 512 * 256 * 2);
  float* Amat  = (float*)bump(p, (size_t)NN * NN * 4);
  float* Vmat  = (float*)bump(p, (size_t)NN * NN * 4);
  float* degv  = (float*)bump(p, (size_t)NN * 4);
  float* pe    = (float*)bump(p, (size_t)NN * 8 * 4);
  float* h0    = (float*)bump(p, (size_t)NTOT * DP * 4);
  float* h1    = (float*)bump(p, (size_t)NTOT * DP * 4);
  _Float16* hf16 = (_Float16*)bump(p, (size_t)NTOT * DP * 2);
  float* qb    = (float*)bump(p, (size_t)NTOT * DP * 4);
  float* kb    = (float*)bump(p, (size_t)NTOT * DP * 4);
  float* vb    = (float*)bump(p, (size_t)NTOT * DP * 4);
  float* hnew  = (float*)bump(p, (size_t)NTOT * DP * 4);
  float* cent  = (float*)bump(p, (size_t)NTOT * DP * 4);
  _Float16* wpad = (_Float16*)bump(p, (size_t)4 * DP * DP * 2);
  float* alpha = (float*)bump(p, (size_t)EBT * 4 * 4);
  float* amax  = (float*)bump(p, (size_t)NTOT * 4 * 4);
  float* denom = (float*)bump(p, (size_t)NTOT * 4 * 4);
  float* meanb = (float*)bump(p, (size_t)BB * DP * 4);
  float* varb  = (float*)bump(p, (size_t)BB * DP * 4);

  const int TPB = 256;
  #define GRD(n) dim3((unsigned)(((n) + TPB - 1) / TPB))

  // ---- 1) positional encoding ----
  k_fill_f32<<<GRD(NN * NN), TPB, 0, stream>>>(Amat, (long)NN * NN, 0.f);
  k_fill_f32<<<GRD(NN), TPB, 0, stream>>>(degv, NN, 0.f);
  k_build_graph<<<GRD(EE), TPB, 0, stream>>>(ei, Amat, degv);
  k_build_L<<<GRD(NN * NN), TPB, 0, stream>>>(Amat, degv);
  k_jacobi<<<1, 256, 0, stream>>>(Amat, Vmat, NN, 12);
  k_pe_extract<<<1, 256, 0, stream>>>(Amat, Vmat, pe);

  // ---- 2) LSTM weight conversion ----
  for (int l = 0; l < 3; ++l)
    for (int d = 0; d < 2; ++d)
      k_f32_to_f16<<<GRD(512 * 128), TPB, 0, stream>>>(
          whh16 + (size_t)(l * 2 + d) * 512 * 128, l_whh[l][d], 512 * 128);
  for (int l = 1; l < 3; ++l)
    for (int d = 0; d < 2; ++d)
      k_f32_to_f16<<<GRD(512 * 256), TPB, 0, stream>>>(
          wih16 + (size_t)((l - 1) * 2 + d) * 512 * 256, l_wih[l][d], 512 * 256);

  // ---- 3) BiLSTM stack ----
  _Float16* lin[3]  = { nullptr, actA, actB };
  _Float16* lout[3] = { actA, actB, actA };
  for (int l = 0; l < 3; ++l) {
    for (int d = 0; d < 2; ++d) {
      const _Float16* wih_h = (l == 0) ? nullptr : wih16 + (size_t)((l - 1) * 2 + d) * 512 * 256;
      k_lstm<<<NTOT / 16, 256, 0, stream>>>(
          (l == 0) ? x : nullptr, lin[l],
          (l == 0) ? l_wih[0][d] : nullptr, wih_h,
          whh16 + (size_t)(l * 2 + d) * 512 * 128,
          l_bih[l][d], l_bhh[l][d],
          lout[l], (l == 0) ? 0 : 256, d /*reverse*/, d * 128 /*dirofs*/);
    }
  }

  // ---- 4) node features ----
  k_node<<<GRD((long)NTOT * DP), TPB, 0, stream>>>(actA, pe, h0);

  // ---- 5) graph transformer layers ----
  float* hcur = h0;
  float* hnext = h1;
  dim3 gblk(32, 4);
  dim3 ggrd((DP / 16 + 3) / 4, NTOT / 16);
  for (int l = 0; l < 3; ++l) {
    _Float16* wq_p = wpad + (size_t)0 * DP * DP;
    _Float16* wk_p = wpad + (size_t)1 * DP * DP;
    _Float16* wv_p = wpad + (size_t)2 * DP * DP;
    _Float16* ws_p = wpad + (size_t)3 * DP * DP;
    k_pad_w<<<GRD((long)DP * DP), TPB, 0, stream>>>(wq_p, g_wq[l]);
    k_pad_w<<<GRD((long)DP * DP), TPB, 0, stream>>>(wk_p, g_wk[l]);
    k_pad_w<<<GRD((long)DP * DP), TPB, 0, stream>>>(wv_p, g_wv[l]);
    k_pad_w<<<GRD((long)DP * DP), TPB, 0, stream>>>(ws_p, g_wskip[l]);
    k_f32_to_f16<<<GRD((long)NTOT * DP), TPB, 0, stream>>>(hf16, hcur, (long)NTOT * DP);

    k_gemm288<<<ggrd, gblk, 0, stream>>>(hf16, wq_p, g_bq[l], DD, qb);
    k_gemm288<<<ggrd, gblk, 0, stream>>>(hf16, wk_p, g_bk[l], DD, kb);
    k_gemm288<<<ggrd, gblk, 0, stream>>>(hf16, wv_p, g_bv[l], DD, vb);
    k_gemm288<<<ggrd, gblk, 0, stream>>>(hf16, ws_p, g_bskip[l], DD, hnew);

    k_fill_f32<<<GRD(NTOT * 4), TPB, 0, stream>>>(amax, NTOT * 4, -1e30f);
    k_fill_f32<<<GRD(NTOT * 4), TPB, 0, stream>>>(denom, NTOT * 4, 0.f);
    k_att1<<<GRD(EBT * 4), TPB, 0, stream>>>(qb, kb, ei, ew, g_we[l], alpha, amax);
    k_att2<<<GRD(EBT * 4), TPB, 0, stream>>>(ei, amax, alpha, denom);
    k_att3<<<GRD(EBT * 4), TPB, 0, stream>>>(ei, ew, g_we[l], vb, alpha, denom, hnew);

    k_gn_mean<<<GRD(BB * DP), TPB, 0, stream>>>(hnew, meanb);
    k_gn_center<<<GRD((long)NTOT * DP), TPB, 0, stream>>>(hnew, meanb, n_ms[l], cent);
    k_gn_var<<<GRD(BB * DP), TPB, 0, stream>>>(cent, varb);
    k_gn_final<<<GRD((long)NTOT * DP), TPB, 0, stream>>>(cent, varb, n_w[l], n_bias[l],
                                                         hcur, hnext);
    float* tmp = hcur; hcur = hnext; hnext = tmp;
  }

  // ---- 6) pooling + classifier ----
  k_pool_cls<<<BB, 256, 0, stream>>>(hcur, cls_w, cls_b, out);
}
